// Block_72146860638363
// MI455X (gfx1250) — matrix operand order repaired and checked by reading, not compiled, not run on hardware
//
#include <hip/hip_runtime.h>
#include <hip/hip_bf16.h>
#include <math.h>

// ---------------- CDNA5 WMMA types ----------------
typedef __attribute__((ext_vector_type(16))) __bf16 v16bf;
typedef __attribute__((ext_vector_type(8)))  float  v8f;

// Problem constants
#define Bc   8
#define Nc   1024
#define DIMc 1536
#define Hc   12
#define Dc   128
#define Mc   64      // random features M
#define Rc   64
#define Sc   5
#define ROWS (Bc * Nc)   // 8192

// =====================================================================
// Fragment loaders for V_WMMA_F32_16X16X32_BF16 (wave32 layouts per ISA)
// A (16x32, row major, lda=K):
//   lane L: row = L&15 ; hi = L>>4
//   elems 0..7  -> K = kBase + hi*8  + e
//   elems 8..15 -> K = kBase + 16 + hi*8 + (e-8)
// B (32x16, from W stored [Ncols,K] row-major => col n of B = row n of W):
//   lane L: col = L&15 ; hi = L>>4 ; elems 0..7 -> K = kBase + hi*8 + e ; elems 8..15 -> K = kBase + 16 + hi*8 + (e-8)
// =====================================================================
__device__ inline v16bf load_a_frag(const float* __restrict__ A, int K,
                                    int rowBase, int kBase, int lane) {
  const int row = rowBase + (lane & 15);
  const int hi  = (lane >> 4) & 1;
  const float* p0 = A + (size_t)row * K + kBase + hi * 8;
  const float* p1 = p0 + 16;
  v16bf a;
#pragma unroll
  for (int i = 0; i < 8; ++i) a[i] = (__bf16)p0[i];
#pragma unroll
  for (int i = 0; i < 8; ++i) a[8 + i] = (__bf16)p1[i];
  return a;
}

__device__ inline v16bf load_b_frag(const float* __restrict__ W, int K,
                                    int colBase, int kBase, int lane) {
  const int col = colBase + (lane & 15);
  const int hi  = (lane >> 4) & 1;
  const float* p = W + (size_t)col * K + kBase + hi * 8;
  v16bf b;
#pragma unroll
  for (int i = 0; i < 8; ++i) { b[i] = (__bf16)p[i]; b[8 + i] = (__bf16)p[16 + i]; }
  return b;
}

// =====================================================================
// GEMM: C[M,Nc] = act(A[M,K] @ W[Nc,K]^T + bias) (+ res)
// act: 0 = none, 1 = exact GELU. bias may be null.
// Block: 256 thr = 8 waves (4 along M x 2 along N).
// Wave tile 32x64 (2x4 WMMA tiles) -> block tile 128x128.
// Mrows % 128 == 0 and K % 32 == 0 for all uses; Ncols guarded.
// =====================================================================
__global__ __launch_bounds__(256) void gemm_wmma_kernel(
    const float* __restrict__ A, const float* __restrict__ W,
    const float* __restrict__ bias, const float* __restrict__ res,
    float* __restrict__ C, int Mrows, int Ncols, int K, int act) {
  const int lane  = threadIdx.x & 31;
  const int wave  = threadIdx.x >> 5;
  const int waveM = wave & 3;
  const int waveN = wave >> 2;
  const int rowBase = blockIdx.y * 128 + waveM * 32;
  const int colBase = blockIdx.x * 128 + waveN * 64;
  if (rowBase >= Mrows || colBase >= Ncols) return;

  v8f acc[2][4];
#pragma unroll
  for (int i = 0; i < 2; ++i)
#pragma unroll
    for (int j = 0; j < 4; ++j)
#pragma unroll
      for (int e = 0; e < 8; ++e) acc[i][j][e] = 0.0f;

  for (int kb = 0; kb < K; kb += 32) {
    v16bf a0 = load_a_frag(A, K, rowBase,      kb, lane);
    v16bf a1 = load_a_frag(A, K, rowBase + 16, kb, lane);
    v16bf b0 = load_b_frag(W, K, colBase,      kb, lane);
    v16bf b1 = load_b_frag(W, K, colBase + 16, kb, lane);
    v16bf b2 = load_b_frag(W, K, colBase + 32, kb, lane);
    v16bf b3 = load_b_frag(W, K, colBase + 48, kb, lane);
    acc[0][0] = __builtin_amdgcn_wmma_f32_16x16x32_bf16(false, a0, false, b0,
                (short)0, acc[0][0], false, false);
    acc[0][1] = __builtin_amdgcn_wmma_f32_16x16x32_bf16(false, a0, false, b1,
                (short)0, acc[0][1], false, false);
    acc[0][2] = __builtin_amdgcn_wmma_f32_16x16x32_bf16(false, a0, false, b2,
                (short)0, acc[0][2], false, false);
    acc[0][3] = __builtin_amdgcn_wmma_f32_16x16x32_bf16(false, a0, false, b3,
                (short)0, acc[0][3], false, false);
    acc[1][0] = __builtin_amdgcn_wmma_f32_16x16x32_bf16(false, a1, false, b0,
                (short)0, acc[1][0], false, false);
    acc[1][1] = __builtin_amdgcn_wmma_f32_16x16x32_bf16(false, a1, false, b1,
                (short)0, acc[1][1], false, false);
    acc[1][2] = __builtin_amdgcn_wmma_f32_16x16x32_bf16(false, a1, false, b2,
                (short)0, acc[1][2], false, false);
    acc[1][3] = __builtin_amdgcn_wmma_f32_16x16x32_bf16(false, a1, false, b3,
                (short)0, acc[1][3], false, false);
  }

  // Epilogue. C/D layout: lane<16 -> M=e, lane>=16 -> M=e+8 ; N = lane&15
  const int mOff = ((lane >> 4) & 1) * 8;
  const int nIdx = lane & 15;
#pragma unroll
  for (int tm = 0; tm < 2; ++tm) {
#pragma unroll
    for (int tn = 0; tn < 4; ++tn) {
      const int col = colBase + tn * 16 + nIdx;
      const float bv = bias ? bias[col] : 0.0f;
#pragma unroll
      for (int e = 0; e < 8; ++e) {
        const int row = rowBase + tm * 16 + e + mOff;
        float v = acc[tm][tn][e] + bv;
        if (act == 1) v = 0.5f * v * (1.0f + erff(v * 0.7071067811865476f));
        if (res) v += res[(size_t)row * Ncols + col];
        C[(size_t)row * Ncols + col] = v;
      }
    }
  }
}

// =====================================================================
// Layernorm (optionally two chained LNs). One block per 1536-wide row.
// =====================================================================
__global__ __launch_bounds__(256) void ln_kernel(
    const float* __restrict__ in, float* __restrict__ out,
    const float* __restrict__ g1, const float* __restrict__ b1,
    const float* __restrict__ g2, const float* __restrict__ b2, int twoLn) {
  __shared__ float red[256];
  const int t = threadIdx.x;
  const float* xr = in + (size_t)blockIdx.x * DIMc;
  float v[6];
  float s = 0.f;
#pragma unroll
  for (int i = 0; i < 6; ++i) { v[i] = xr[t + i * 256]; s += v[i]; }
  red[t] = s; __syncthreads();
  for (int w = 128; w > 0; w >>= 1) { if (t < w) red[t] += red[t + w]; __syncthreads(); }
  float mu = red[0] * (1.f / DIMc);
  __syncthreads();
  s = 0.f;
#pragma unroll
  for (int i = 0; i < 6; ++i) { float d = v[i] - mu; s += d * d; }
  red[t] = s; __syncthreads();
  for (int w = 128; w > 0; w >>= 1) { if (t < w) red[t] += red[t + w]; __syncthreads(); }
  float rs = rsqrtf(red[0] * (1.f / DIMc) + 1e-5f);
  __syncthreads();
#pragma unroll
  for (int i = 0; i < 6; ++i) v[i] = (v[i] - mu) * rs * g1[t + i * 256] + b1[t + i * 256];

  if (twoLn) {
    s = 0.f;
#pragma unroll
    for (int i = 0; i < 6; ++i) s += v[i];
    red[t] = s; __syncthreads();
    for (int w = 128; w > 0; w >>= 1) { if (t < w) red[t] += red[t + w]; __syncthreads(); }
    mu = red[0] * (1.f / DIMc);
    __syncthreads();
    s = 0.f;
#pragma unroll
    for (int i = 0; i < 6; ++i) { float d = v[i] - mu; s += d * d; }
    red[t] = s; __syncthreads();
    for (int w = 128; w > 0; w >>= 1) { if (t < w) red[t] += red[t + w]; __syncthreads(); }
    rs = rsqrtf(red[0] * (1.f / DIMc) + 1e-5f);
    __syncthreads();
#pragma unroll
    for (int i = 0; i < 6; ++i) v[i] = (v[i] - mu) * rs * g2[t + i * 256] + b2[t + i * 256];
  }
  float* orow = out + (size_t)blockIdx.x * DIMc;
#pragma unroll
  for (int i = 0; i < 6; ++i) orow[t + i * 256] = v[i];
}

// =====================================================================
// SPE feature kernel: per (n, h, q/k) block builds qh/kh rows for all b.
// =====================================================================
__global__ __launch_bounds__(256) void spe_features_kernel(
    const float* __restrict__ qkv, const float* __restrict__ freqs,
    const float* __restrict__ offs, const float* __restrict__ gains,
    const float* __restrict__ gate, const float* __restrict__ znoise,
    const float* __restrict__ gnoise, float* __restrict__ qh,
    float* __restrict__ kh) {
  const int n = blockIdx.x, h = blockIdx.y;
  const int isQ = (blockIdx.z == 0);
  __shared__ float omg[Dc * 10];       // omega * softplus(gains)
  __shared__ float qb[Dc * 65];        // padded stride 65 (bank-conflict free)
  __shared__ float coefA[Dc];
  __shared__ float coefG[Dc];
  __shared__ float qrow[Bc * Dc];
  const int t = threadIdx.x;
  const float TWO_PI = 6.283185307179586f;
  const float INV_SCALE_R4 = 1.0f / (9.51365692f * 2.8284271247461903f); // 1/((R*D)^.25 * R^.25)

  if (t < Dc) {
    float gt = 1.f / (1.f + expf(-gate[h * Dc + t]));
    coefA[t] = sqrtf(1.f - gt) * INV_SCALE_R4;
    coefG[t] = sqrtf(gt) * INV_SCALE_R4;
  }
  for (int i = t; i < Dc * 10; i += 256) {
    const int d = i / 10, s2 = i % 10, s = s2 >> 1;
    const int gi = (h * Dc + d) * Sc + s;
    const float f  = 0.5f / (1.f + expf(-freqs[gi]));
    float ph = TWO_PI * f * (float)n;
    if (isQ) ph += offs[gi];
    const float val = (s2 & 1) ? sinf(ph) : cosf(ph);
    const float gx = gains[gi];
    const float sp = (gx > 20.f) ? gx : log1pf(expf(gx));
    omg[i] = val * sp;
  }
  for (int i = t; i < Bc * Dc; i += 256) {
    const int b = i >> 7, d = i & 127;
    qrow[i] = qkv[((size_t)(b * Nc + n)) * (3 * DIMc) + (isQ ? 0 : DIMc) + h * Dc + d];
  }
  __syncthreads();
  for (int i = t; i < Dc * Rc; i += 256) {
    const int d = i >> 6, r = i & 63;
    const float* zp = znoise + ((size_t)(h * Dc + d) * 10) * Rc + r;
    float a = 0.f;
#pragma unroll
    for (int s2 = 0; s2 < 10; ++s2) a += omg[d * 10 + s2] * zp[s2 * Rc];
    qb[d * 65 + r] = a * coefA[d];
  }
  __syncthreads();
  float* out = isQ ? qh : kh;
  for (int i = t; i < Bc * 128; i += 256) {
    const int b = i >> 7, r = i & 127;
    float a = 0.f;
    if (r < Rc) {
#pragma unroll 8
      for (int d = 0; d < Dc; ++d) a += qrow[b * Dc + d] * qb[d * 65 + r];
    } else {
      const int rr = r - Rc;
#pragma unroll 8
      for (int d = 0; d < Dc; ++d)
        a += qrow[b * Dc + d] * coefG[d] * gnoise[(h * Dc + d) * Rc + rr];
    }
    out[(((size_t)b * Hc + h) * Nc + n) * 128 + r] = a;
  }
}

// ---------------- sortable float <-> uint for global atomic max ---------
__device__ inline unsigned fsort(float f) {
  unsigned u = __float_as_uint(f);
  return (u & 0x80000000u) ? ~u : (u | 0x80000000u);
}
__device__ inline float funsort(unsigned u) {
  return (u & 0x80000000u) ? __uint_as_float(u ^ 0x80000000u)
                           : __uint_as_float(~u);
}

// =====================================================================
// phi post-pass for Q. qp currently holds raw qh@w^T; apply dn, diag,
// per-row max, exp. One block (64 thr) per (b,h,n).
// =====================================================================
__global__ __launch_bounds__(64) void phi_q_post_kernel(
    const float* __restrict__ qh, float* __restrict__ qp) {
  __shared__ float red[64];
  const int t = threadIdx.x;
  const size_t row = blockIdx.x;
  const float* xr = qh + row * 128;
  red[t] = xr[t] * xr[t] + xr[t + 64] * xr[t + 64];
  __syncthreads();
  for (int s = 32; s > 0; s >>= 1) { if (t < s) red[t] += red[t + s]; __syncthreads(); }
  const float dn = 0.29730177875068026f;            // 128^-0.25
  const float diag = 0.5f * dn * dn * red[0];
  __syncthreads();
  const float dd = dn * qp[row * Mc + t];
  red[t] = dd; __syncthreads();
  for (int s = 32; s > 0; s >>= 1) { if (t < s) red[t] = fmaxf(red[t], red[t + s]); __syncthreads(); }
  const float mx = red[0];
  qp[row * Mc + t] = 0.125f * (expf(dd - diag - mx) + 1e-4f);  // ratio = M^-0.5
}

// phi post-pass for K, pass A: kp holds raw kh@w^T. Store dd - diag,
// accumulate global per-(b,h) max of dd.
__global__ __launch_bounds__(64) void phi_k_post_kernel(
    const float* __restrict__ kh, float* __restrict__ kp,
    unsigned* __restrict__ kmax) {
  __shared__ float red[64];
  const int t = threadIdx.x;
  const size_t row = blockIdx.x;
  const float* xr = kh + row * 128;
  red[t] = xr[t] * xr[t] + xr[t + 64] * xr[t + 64];
  __syncthreads();
  for (int s = 32; s > 0; s >>= 1) { if (t < s) red[t] += red[t + s]; __syncthreads(); }
  const float dn = 0.29730177875068026f;
  const float diag = 0.5f * dn * dn * red[0];
  __syncthreads();
  const float dd = dn * kp[row * Mc + t];
  kp[row * Mc + t] = dd - diag;
  red[t] = dd; __syncthreads();
  for (int s = 32; s > 0; s >>= 1) { if (t < s) red[t] = fmaxf(red[t], red[t + s]); __syncthreads(); }
  if (t == 0) atomicMax(&kmax[row / Nc], fsort(red[0]));
}

// phi for K, pass B: exponentiate with the global max
__global__ __launch_bounds__(256) void phi_k_exp_kernel(
    float* __restrict__ kp, const unsigned* __restrict__ kmax, size_t total) {
  const size_t i = (size_t)blockIdx.x * 256 + threadIdx.x;
  if (i >= total) return;
  const size_t bh = i / ((size_t)Nc * Mc);
  const float mx = funsort(kmax[bh]);
  kp[i] = 0.125f * (expf(kp[i] - mx) + 1e-4f);
}

// ksum[b,h,m] = sum_n kp[b,h,n,m]
__global__ __launch_bounds__(64) void ksum_kernel(const float* __restrict__ kp,
                                                  float* __restrict__ ksum) {
  const int bh = blockIdx.x, m = threadIdx.x;
  const float* p = kp + (size_t)bh * Nc * Mc + m;
  float s = 0.f;
  for (int n = 0; n < Nc; ++n) s += p[(size_t)n * Mc];
  ksum[bh * Mc + m] = s;
}

// ctx[b,h,m,d] = sum_n kp[b,h,n,m] * v[b,h,n,d]
__global__ __launch_bounds__(128) void ctx_kernel(const float* __restrict__ kp,
                                                  const float* __restrict__ qkv,
                                                  float* __restrict__ ctx) {
  const int m = blockIdx.x, h = blockIdx.y, b = blockIdx.z, d = threadIdx.x;
  const float* kpp = kp + ((size_t)(b * Hc + h) * Nc) * Mc + m;
  const float* vp  = qkv + ((size_t)(b * Nc)) * (3 * DIMc) + 2 * DIMc + h * Dc + d;
  float acc = 0.f;
  for (int n = 0; n < Nc; ++n)
    acc += kpp[(size_t)n * Mc] * vp[(size_t)n * (3 * DIMc)];
  ctx[(((size_t)(b * Hc + h)) * Mc + m) * Dc + d] = acc;
}

// y[b,n,h*D+d] = dinv * sum_m qp[b,h,n,m]*ctx[b,h,m,d] / sqrt(R)
__global__ __launch_bounds__(128) void y_kernel(
    const float* __restrict__ qp, const float* __restrict__ ksum,
    const float* __restrict__ ctx, float* __restrict__ ybuf) {
  __shared__ float qs[Mc];
  __shared__ float red[Mc];
  const int n = blockIdx.x, h = blockIdx.y, b = blockIdx.z, t = threadIdx.x;
  const size_t row = (size_t)(b * Hc + h) * Nc + n;
  if (t < Mc) {
    const float qv = qp[row * Mc + t];
    qs[t] = qv;
    red[t] = qv * ksum[(b * Hc + h) * Mc + t];
  }
  __syncthreads();
  for (int s = 32; s > 0; s >>= 1) { if (t < s) red[t] += red[t + s]; __syncthreads(); }
  const float dinv = 1.f / red[0];
  const float* cx = ctx + ((size_t)(b * Hc + h) * Mc) * Dc + t;
  float acc = 0.f;
#pragma unroll 8
  for (int m = 0; m < Mc; ++m) acc += qs[m] * cx[(size_t)m * Dc];
  ybuf[((size_t)(b * Nc) + n) * DIMc + h * Dc + t] = acc * dinv * 0.125f; // 1/sqrt(R)
}

// d_out = x + v_skip + proj_out
__global__ __launch_bounds__(256) void residual_kernel(
    const float* __restrict__ x, const float* __restrict__ qkv,
    const float* __restrict__ proj, float* __restrict__ out, size_t total) {
  const size_t i = (size_t)blockIdx.x * 256 + threadIdx.x;
  if (i >= total) return;
  const size_t row = i / DIMc;
  const size_t c = i - row * DIMc;
  out[i] = x[i] + qkv[row * (3 * DIMc) + 2 * DIMc + c] + proj[i];
}

// =====================================================================
extern "C" void kernel_launch(void* const* d_in, const int* in_sizes, int n_in,
                              void* d_out, int out_size, void* d_ws,
                              size_t ws_size, hipStream_t stream) {
  const float* x      = (const float*)d_in[0];
  const float* ln0_g  = (const float*)d_in[1];
  const float* ln0_b  = (const float*)d_in[2];
  const float* lna_g  = (const float*)d_in[3];
  const float* lna_b  = (const float*)d_in[4];
  const float* Wqkv   = (const float*)d_in[5];
  const float* bqkv   = (const float*)d_in[6];
  const float* Wproj  = (const float*)d_in[7];
  const float* bproj  = (const float*)d_in[8];
  const float* w      = (const float*)d_in[9];
  const float* freqs  = (const float*)d_in[10];
  const float* offs   = (const float*)d_in[11];
  const float* gains  = (const float*)d_in[12];
  const float* gate   = (const float*)d_in[13];
  const float* znoise = (const float*)d_in[14];
  const float* gnoise = (const float*)d_in[15];
  const float* ln2_g  = (const float*)d_in[16];
  const float* ln2_b  = (const float*)d_in[17];
  const float* W1     = (const float*)d_in[18];
  const float* b1     = (const float*)d_in[19];
  const float* W2     = (const float*)d_in[20];
  const float* b2     = (const float*)d_in[21];
  float* out = (float*)d_out;

  float* ws = (float*)d_ws;
  size_t o = 0;
  float* h    = ws + o; o += (size_t)ROWS * DIMc;          // 12.6M
  float* qh   = ws + o; o += (size_t)Bc * Hc * Nc * 128;   // 12.6M
  float* kh   = ws + o; o += (size_t)Bc * Hc * Nc * 128;   // 12.6M
  float* qp   = ws + o; o += (size_t)Bc * Hc * Nc * Mc;    // 6.3M
  float* kp   = ws + o; o += (size_t)Bc * Hc * Nc * Mc;    // 6.3M
  float* ctx  = ws + o; o += (size_t)Bc * Hc * Mc * Dc;    // 0.79M
  float* ksum = ws + o; o += (size_t)Bc * Hc * Mc;
  unsigned* kmax = (unsigned*)(ws + o); o += 128;          // B*H = 96 (padded)
  float* ybuf = ws + o; o += (size_t)ROWS * DIMc;          // 12.6M
  float* proj = ws + o; o += (size_t)ROWS * DIMc;          // 12.6M
  float* qkv  = ws + o; o += (size_t)ROWS * 3 * DIMc;      // 37.7M
  // Reuse (dead after residual): h2 in qh's slot; h2g spans kh..proj (51.1M >= 50.3M)
  float* h2  = qh;
  float* h2g = kh;

  const int PHIROWS = Bc * Hc * Nc;   // 98304 (divisible by 128)

  // 1. h = LN_a(LN_0(x))
  ln_kernel<<<ROWS, 256, 0, stream>>>(x, h, ln0_g, ln0_b, lna_g, lna_b, 1);
  // 2. qkv = h @ Wqkv^T + bqkv         (WMMA)
  gemm_wmma_kernel<<<dim3(3 * DIMc / 128, ROWS / 128), 256, 0, stream>>>(
      h, Wqkv, bqkv, nullptr, qkv, ROWS, 3 * DIMc, DIMc, 0);
  // 3. SPE features -> qh, kh
  spe_features_kernel<<<dim3(Nc, Hc, 2), 256, 0, stream>>>(
      qkv, freqs, offs, gains, gate, znoise, gnoise, qh, kh);
  // 4. raw dd projections via WMMA:  qp = qh @ w^T ; kp = kh @ w^T
  gemm_wmma_kernel<<<dim3(1, PHIROWS / 128), 256, 0, stream>>>(
      qh, w, nullptr, nullptr, qp, PHIROWS, Mc, 128, 0);
  gemm_wmma_kernel<<<dim3(1, PHIROWS / 128), 256, 0, stream>>>(
      kh, w, nullptr, nullptr, kp, PHIROWS, Mc, 128, 0);
  // 5. phi epilogues
  hipMemsetAsync(kmax, 0, 128 * sizeof(unsigned), stream);
  phi_q_post_kernel<<<PHIROWS, 64, 0, stream>>>(qh, qp);
  phi_k_post_kernel<<<PHIROWS, 64, 0, stream>>>(kh, kp, kmax);
  {
    const size_t total = (size_t)PHIROWS * Mc;
    phi_k_exp_kernel<<<(total + 255) / 256, 256, 0, stream>>>(kp, kmax, total);
  }
  // 6. ksum
  ksum_kernel<<<Bc * Hc, 64, 0, stream>>>(kp, ksum);
  // 7. ctx = kp^T @ v
  ctx_kernel<<<dim3(Mc, Hc, Bc), 128, 0, stream>>>(kp, qkv, ctx);
  // 8. y
  y_kernel<<<dim3(Nc, Hc, Bc), 128, 0, stream>>>(qp, ksum, ctx, ybuf);
  // 9. proj = y @ Wproj^T + bproj      (WMMA)
  gemm_wmma_kernel<<<dim3(DIMc / 128, ROWS / 128), 256, 0, stream>>>(
      ybuf, Wproj, bproj, nullptr, proj, ROWS, DIMc, DIMc, 0);
  // 10. out = x + v_skip + proj
  {
    const size_t total = (size_t)ROWS * DIMc;
    residual_kernel<<<(total + 255) / 256, 256, 0, stream>>>(x, qkv, proj, out, total);
  }
  // 11. h2 = LN2(out)
  ln_kernel<<<ROWS, 256, 0, stream>>>(out, h2, ln2_g, ln2_b, nullptr, nullptr, 0);
  // 12. h2g = gelu(h2 @ W1^T + b1)     (WMMA, fused GELU)
  gemm_wmma_kernel<<<dim3(4 * DIMc / 128, ROWS / 128), 256, 0, stream>>>(
      h2, W1, b1, nullptr, h2g, ROWS, 4 * DIMc, DIMc, 1);
  // 13. out += h2g @ W2^T + b2         (WMMA, fused residual)
  gemm_wmma_kernel<<<dim3(DIMc / 128, ROWS / 128), 256, 0, stream>>>(
      h2g, W2, b2, out, out, ROWS, DIMc, 4 * DIMc, 0);
}
